// YOLOLoss_83459804496167
// MI455X (gfx1250) — compile-verified, matrix operand order checked by reading, not run on hardware
//
#include <hip/hip_runtime.h>
#include <hip/hip_bf16.h>
#include <hip/hip_fp16.h>

// ---------------------------------------------------------------------------
// YOLOv3 loss on gfx1250 (MI455X, wave32).
//   B=16, A=3, C=80, INPUT_DIM=640, grids 80/40/20, NT=256 targets.
// Dense work  : only channel 4 (objectness) of every cell -> 403,200 elems.
// Sparse work : <=256 object cells/scale, 85 channels each + noobj correction.
// Block reduction of 256 f32 partials done with V_WMMA_F32_16X16X4_F32.
// ---------------------------------------------------------------------------

typedef __attribute__((ext_vector_type(2)))  float    v2f;
typedef __attribute__((ext_vector_type(8)))  float    v8f;
typedef __attribute__((ext_vector_type(16))) _Float16 v16h;

#define YB 16
#define YA 3
#define YC 80
#define YDIM 640
#define NT 256
#define N0 (16*3*80*80)   /* 307200 obj cells, scale 0 */
#define N1 (16*3*40*40)   /*  76800 obj cells, scale 1 */
#define N2 (16*3*20*20)   /*  19200 obj cells, scale 2 */
#define NOBJ_TOTAL (N0+N1+N2)  /* 403200 */

// -clog(1-sigmoid(x)) == min(softplus(x), 100)   (torch-style BCE clamp)
__device__ __forceinline__ float softplus_clamped(float x) {
    float sp = (x > 0.f) ? (x + log1pf(__expf(-x))) : log1pf(__expf(x));
    return fminf(sp, 100.f);
}
__device__ __forceinline__ float sigmoidf(float x) {
    return 1.f / (1.f + __expf(-x));
}

// ---------------------------------------------------------------------------
// Reduce 256 f32 partials (one per thread of a 256-thread block) to a scalar.
// Returns the total on threadIdx.x == 0 (other threads' values unspecified). Uses WMMA:
//   D = A(16x4) * ones(4x16) + C  =>  D[i][j] += rowsum_i ; chained 4x over
//   the 4 chunks of 64 values, every column of D ends up holding all 16
//   row-sums.  C/D layout: lane L<16: VGPR v = D[v][L]; lane>=16: D[v+8][L-16].
//   => lane0 holds D[0..7][0], lane16 holds D[8..15][0]; add + shfl(16).
// ---------------------------------------------------------------------------
__device__ float blockReduce256(float v, float* s256) {
    const int tid = threadIdx.x;
    s256[tid] = v;
    __syncthreads();
    float result = 0.f;
#if __has_builtin(__builtin_amdgcn_wmma_f32_16x16x4_f32)
    if (tid < 32) {                       // whole wave 0: EXEC all-1s for WMMA
        v8f acc = {};
        v2f bONE; bONE[0] = 1.f; bONE[1] = 1.f;   // ones: layout-agnostic
        #pragma unroll
        for (int c = 0; c < 4; ++c) {
            const float* base = s256 + c * 64;
            // f32 A 16x4 layout: lanes 0-15 -> M=lane, K=0,1 ; lanes 16-31 -> K=2,3
            v2f a;
            if (tid < 16) { a[0] = base[tid*4 + 0];      a[1] = base[tid*4 + 1]; }
            else          { a[0] = base[(tid-16)*4 + 2]; a[1] = base[(tid-16)*4 + 3]; }
            acc = __builtin_amdgcn_wmma_f32_16x16x4_f32(
                false, a, false, bONE, (short)0, acc, false, false);
        }
        float s = acc[0]+acc[1]+acc[2]+acc[3]+acc[4]+acc[5]+acc[6]+acc[7];
        s += __shfl(s, 16, 32);           // lane0: rows0-7 + rows8-15 of col 0
        result = s;                       // grand total valid on lane 0
    }
#elif __has_builtin(__builtin_amdgcn_wmma_f32_16x16x32_f16)
    if (tid < 32) {
        // 16-bit A 16x32 layout: lanes0-15: V0..3 = K0..7, V4..7 = K16..23;
        //                        lanes16-31: V0..3 = K8..15, V4..7 = K24..31.
        // Put the 256 values in K=0..15 (row m = vals[m*16 + k]); pad K>=16 w/ 0.
        const int m = (tid < 16) ? tid : (tid - 16);
        v16h a;
        #pragma unroll
        for (int vv = 0; vv < 8; ++vv) {
            bool pad = (vv >= 4);
            int  k0  = (tid < 16) ? (2*vv) : (8 + 2*vv);
            a[2*vv]   = pad ? (_Float16)0.f : (_Float16)s256[m*16 + k0];
            a[2*vv+1] = pad ? (_Float16)0.f : (_Float16)s256[m*16 + k0 + 1];
        }
        v16h bONE;
        #pragma unroll
        for (int i = 0; i < 16; ++i) bONE[i] = (_Float16)1.f;
        v8f acc = {};
        acc = __builtin_amdgcn_wmma_f32_16x16x32_f16(
            false, a, false, bONE, (short)0, acc, false, false);
        float s = acc[0]+acc[1]+acc[2]+acc[3]+acc[4]+acc[5]+acc[6]+acc[7];
        s += __shfl(s, 16, 32);
        result = s;
    }
#else
    for (int off = 128; off > 0; off >>= 1) {
        if (tid < off) s256[tid] += s256[tid + off];
        __syncthreads();
    }
    result = (tid == 0) ? s256[0] : 0.f;
#endif
    return result;
}

__global__ void yolo_init(float* out) { out[0] = 0.f; }

// ---------------------------------------------------------------------------
// Dense pass: 0.5 * sum over ALL cells of min(softplus(obj_logit),100).
// (Object cells get a correction in the sparse kernel.)
// ---------------------------------------------------------------------------
__global__ void yolo_dense_noobj(const float* __restrict__ p0,
                                 const float* __restrict__ p1,
                                 const float* __restrict__ p2,
                                 float* __restrict__ out) {
    __shared__ float s256[256];
    float local = 0.f;
    for (int i = blockIdx.x * blockDim.x + threadIdx.x; i < NOBJ_TOTAL;
         i += gridDim.x * blockDim.x) {
        const float* base; int cell;
        if (i < N0)            { base = p0; cell = i; }
        else if (i < N0 + N1)  { base = p1; cell = i - N0; }
        else                   { base = p2; cell = i - (N0 + N1); }
        float x = base[(size_t)cell * 85 + 4];     // objectness logit
        local += 0.5f * softplus_clamped(x);
    }
    float tot = blockReduce256(local, s256);
    if (threadIdx.x == 0) atomicAdd(out, tot);
}

// ---------------------------------------------------------------------------
// Sparse pass: one block, thread t <-> target t. For each scale: build target
// (best anchor by IoU, cell, tx/ty/tw/th), dedupe cells (last writer wins,
// mimicking scatter .set), then per winning cell add x/y/w/h MSE, obj BCE,
// class BCE, and remove the dense pass's noobj term at that cell.
// ---------------------------------------------------------------------------
__global__ void yolo_sparse(const float* __restrict__ p0,
                            const float* __restrict__ p1,
                            const float* __restrict__ p2,
                            const float* __restrict__ anchors,  // [3][3][2] grid units
                            const float* __restrict__ targets,  // [256][6]
                            float* __restrict__ out) {
    __shared__ int   keys[NT];
    __shared__ float s256[256];
    const int tid = threadIdx.x;

    const float tb  = targets[tid*6 + 0];
    const float tcl = targets[tid*6 + 1];
    const float txn = targets[tid*6 + 2];
    const float tyn = targets[tid*6 + 3];
    const float twn = targets[tid*6 + 4];
    const float thn = targets[tid*6 + 5];
    const int bi = (int)tb;
    const int ci = (int)tcl;

    const float* preds[3] = { p0, p1, p2 };
    const int    gsz[3]   = { 80, 40, 20 };

    float local = 0.f;
    for (int s = 0; s < 3; ++s) {
        const int   g      = gsz[s];
        const float stride = (float)YDIM / (float)g;
        const float x = txn * g, y = tyn * g;
        const float w = twn * (float)YDIM, h = thn * (float)YDIM;
        const bool valid = (x >= 0.f) && (y >= 0.f) &&
                           (x <= (float)(g - 1)) && (y <= (float)(g - 1));
        const int gx = (int)floorf(x), gy = (int)floorf(y);

        // Best anchor by IoU of (w,h) vs scaled anchors (first max wins, like argmax).
        float best_iou = -1.f; float sawb = 1.f, sahb = 1.f; int best = 0;
        #pragma unroll
        for (int a = 0; a < 3; ++a) {
            const float saw = anchors[(s*3 + a)*2 + 0] * stride;
            const float sah = anchors[(s*3 + a)*2 + 1] * stride;
            const float inter = fminf(w, saw) * fminf(h, sah);
            const float iou   = inter / (w*h + saw*sah - inter + 1e-16f);
            if (iou > best_iou) { best_iou = iou; best = a; sawb = saw; sahb = sah; }
        }
        const float tx = x - (float)gx, ty = y - (float)gy;
        const float tw = logf(w / sawb + 1e-16f);
        const float th = logf(h / sahb + 1e-16f);

        // Flat cell index in [B,A,g,g] == key; -1 for dropped targets.
        const int key = valid ? (((bi*YA + best)*g + gy)*g + gx) : -1;
        __syncthreads();            // keys[] reuse across scales
        keys[tid] = key;
        __syncthreads();

        bool alive = valid;
        if (alive) {                // last writer wins (scatter-set semantics)
            for (int j = tid + 1; j < NT; ++j)
                if (keys[j] == key) { alive = false; break; }
        }
        if (alive) {
            const float* p = preds[s] + (size_t)key * 85;
            float l = 0.f;
            const float sx = sigmoidf(p[0]); l += (sx - tx) * (sx - tx);
            const float sy = sigmoidf(p[1]); l += (sy - ty) * (sy - ty);
            l += (p[2] - tw) * (p[2] - tw);
            l += (p[3] - th) * (p[3] - th);
            l += softplus_clamped(-p[4]);          // obj BCE at object cell
            l -= 0.5f * softplus_clamped(p[4]);    // undo dense noobj term here
            #pragma unroll 4
            for (int c = 0; c < YC; ++c) {
                const float xc = p[5 + c];
                l += (c == ci) ? softplus_clamped(-xc) : softplus_clamped(xc);
            }
            local += l;
        }
    }
    __syncthreads();
    float tot = blockReduce256(local, s256);
    if (tid == 0) atomicAdd(out, tot);
}

extern "C" void kernel_launch(void* const* d_in, const int* in_sizes, int n_in,
                              void* d_out, int out_size, void* d_ws, size_t ws_size,
                              hipStream_t stream) {
    const float* p0      = (const float*)d_in[0];  // [16,3,80,80,85]
    const float* p1      = (const float*)d_in[1];  // [16,3,40,40,85]
    const float* p2      = (const float*)d_in[2];  // [16,3,20,20,85]
    const float* anchors = (const float*)d_in[3];  // [3,3,2] grid units
    const float* targets = (const float*)d_in[4];  // [256,6]
    float* out = (float*)d_out;

    yolo_init<<<1, 1, 0, stream>>>(out);
    yolo_dense_noobj<<<NOBJ_TOTAL / 256, 256, 0, stream>>>(p0, p1, p2, out);
    yolo_sparse<<<1, 256, 0, stream>>>(p0, p1, p2, anchors, targets, out);
}
